// EncoderLayer_76759655514827
// MI455X (gfx1250) — matrix operand order repaired and checked by reading, not compiled, not run on hardware
//
#include <hip/hip_runtime.h>
#include <hip/hip_bf16.h>
#include <math.h>

// ---------------- problem constants ----------------
#define BATCH 4
#define SEQ   1024
#define DIM   1024
#define NH    16
#define DKH   64
#define NE    8
#define DFF   2048
#define TOK   (BATCH * SEQ)   // 4096

typedef __attribute__((ext_vector_type(16))) __bf16 v16bf;
typedef __attribute__((ext_vector_type(8)))  float  v8f;

union FragA {         // 16 bf16 = 8 VGPRs (WMMA A or B operand)
    v16bf v;
    __bf16 h[16];
    uint4  q[2];
};

// pack 4 fp32 -> 4 bf16 (v_cvt_pk_bf16_f32 x2) and store 8B to LDS
__device__ __forceinline__ void cvt_store8(__bf16* dst, const float4 f) {
    union { uint2 u; __bf16 h[4]; } pk;
    pk.h[0] = (__bf16)f.x; pk.h[1] = (__bf16)f.y;
    pk.h[2] = (__bf16)f.z; pk.h[3] = (__bf16)f.w;
    *(uint2*)dst = pk.u;
}
// load 8 contiguous fp32 (2x b128) -> 8 bf16
__device__ __forceinline__ void load8_cvt(__bf16* dst, const float* __restrict__ p) {
    #pragma unroll
    for (int i = 0; i < 2; ++i) {
        const float4 f = *(const float4*)(p + i * 4);
        dst[i*4+0] = (__bf16)f.x; dst[i*4+1] = (__bf16)f.y;
        dst[i*4+2] = (__bf16)f.z; dst[i*4+3] = (__bf16)f.w;
    }
}
// load 16 contiguous fp32 (4x b128) -> 16 bf16
__device__ __forceinline__ void load16_cvt(FragA& fr, const float* __restrict__ p) {
    load8_cvt(&fr.h[0], p);
    load8_cvt(&fr.h[8], p + 8);
}

// =====================================================================
// Generic bf16 WMMA GEMM:  C[M,N] = epilogue( A[M,K] * B[N,K]^T )
//   mode 0: C = v
//   mode 1: C = gelu(v)
//   mode 2: o = resid + v*rs; C = o; C2 = o      (attention residual)
//   mode 3: C += gate[m*8] * rs * v              (MoE weighted accumulate)
// Block tile 64x128x32, 8 waves (2x4), wave tile 32x32 (2x2 WMMA).
// Double-buffered LDS with register staging.
// =====================================================================
__global__ __launch_bounds__(256)
void gemm_bf16_kernel(const float* __restrict__ A, const float* __restrict__ Bm,
                      float* __restrict__ C, float* __restrict__ C2,
                      const float* __restrict__ resid, const float* __restrict__ gate,
                      const float* __restrict__ rs_ptr,
                      int M, int N, int K, int mode)
{
    __shared__ __align__(16) __bf16 As[2][64][40];    // 80B rows, 16B aligned
    __shared__ __align__(16) __bf16 Bs[2][128][40];

    const int tid  = threadIdx.x;
    const int wave = tid >> 5, lane = tid & 31;
    const int lo   = lane & 15, hi = lane >> 4;
    const int wm   = wave & 1,  wn = wave >> 1;          // 2 x 4 wave grid
    const int m0   = blockIdx.y * 64, n0 = blockIdx.x * 128;

    // staging ownership: thread -> rows {sr, sr+32, ...}, 4-float column sc
    const int sr = tid >> 3;            // 0..31
    const int sc = (tid & 7) << 2;      // 0,4,...,28  (floats)
    const float* Abase = A  + (size_t)m0 * K + sc;
    const float* Bbase = Bm + (size_t)n0 * K + sc;

    float4 ra[2], rb[4];
    v8f acc[2][2] = {};
    const int nk = K >> 5;

    // ---- preload K-tile 0 into buffer 0 ----
    #pragma unroll
    for (int it = 0; it < 2; ++it) ra[it] = *(const float4*)(Abase + (sr + it*32) * K);
    #pragma unroll
    for (int it = 0; it < 4; ++it) rb[it] = *(const float4*)(Bbase + (sr + it*32) * K);
    #pragma unroll
    for (int it = 0; it < 2; ++it) cvt_store8(&As[0][sr + it*32][sc], ra[it]);
    #pragma unroll
    for (int it = 0; it < 4; ++it) cvt_store8(&Bs[0][sr + it*32][sc], rb[it]);
    __syncthreads();

    for (int kb = 0; kb < nk; ++kb) {
        const int cur = kb & 1, nxt = cur ^ 1;
        // issue next-tile global loads (latency hidden behind WMMA)
        if (kb + 1 < nk) {
            const int kOff = (kb + 1) << 5;
            #pragma unroll
            for (int it = 0; it < 2; ++it) ra[it] = *(const float4*)(Abase + (sr + it*32) * K + kOff);
            #pragma unroll
            for (int it = 0; it < 4; ++it) rb[it] = *(const float4*)(Bbase + (sr + it*32) * K + kOff);
            if (kb + 2 < nk) {   // global_prefetch_b8 for tile after next
                __builtin_prefetch(Abase + sr * K + ((kb + 2) << 5), 0, 1);
                __builtin_prefetch(Bbase + sr * K + ((kb + 2) << 5), 0, 1);
            }
        }
        // fragments from current LDS buffer
        FragA a[2], b[2];
        #pragma unroll
        for (int i = 0; i < 2; ++i) {   // A 16x32: half-wave K split {0..7,16..23}/{8..15,24..31}
            const __bf16* rp = &As[cur][wm * 32 + i * 16 + lo][0];
            a[i].q[0] = *(const uint4*)(rp + hi * 8);
            a[i].q[1] = *(const uint4*)(rp + 16 + hi * 8);
        }
        #pragma unroll
        for (int j = 0; j < 2; ++j) {   // B 32x16: lane n=lo, half-wave K split {0..7,16..23}/{8..15,24..31}
            const __bf16* rp = &Bs[cur][wn * 32 + j * 16 + lo][0];
            b[j].q[0] = *(const uint4*)(rp + hi * 8);
            b[j].q[1] = *(const uint4*)(rp + 16 + hi * 8);
        }
        #pragma unroll
        for (int i = 0; i < 2; ++i)
            #pragma unroll
            for (int j = 0; j < 2; ++j)
                acc[i][j] = __builtin_amdgcn_wmma_f32_16x16x32_bf16(
                    false, a[i].v, false, b[j].v, (short)0, acc[i][j], false, false);
        // commit next tile to the other LDS buffer
        if (kb + 1 < nk) {
            #pragma unroll
            for (int it = 0; it < 2; ++it) cvt_store8(&As[nxt][sr + it*32][sc], ra[it]);
            #pragma unroll
            for (int it = 0; it < 4; ++it) cvt_store8(&Bs[nxt][sr + it*32][sc], rb[it]);
        }
        __syncthreads();
    }

    const float rs = rs_ptr ? rs_ptr[0] : 1.0f;
    #pragma unroll
    for (int i = 0; i < 2; ++i)
        #pragma unroll
        for (int j = 0; j < 2; ++j)
            #pragma unroll
            for (int r = 0; r < 8; ++r) {
                const int m = m0 + wm * 32 + i * 16 + hi * 8 + r;  // D row = r + 8*(lane>=16)
                const int n = n0 + wn * 32 + j * 16 + lo;          // D col = lane%16
                const size_t idx = (size_t)m * N + n;
                float v = acc[i][j][r];
                if (mode == 0) {
                    C[idx] = v;
                } else if (mode == 1) {
                    C[idx] = 0.5f * v * (1.0f + erff(v * 0.70710678118654752f));
                } else if (mode == 2) {
                    float o = resid[idx] + v * rs;
                    C[idx] = o;
                    C2[idx] = o;
                } else {
                    C[idx] += gate[(size_t)m * 8] * rs * v;
                }
            }
}

// =====================================================================
// LayerNorm over last dim (D=1024), one block (256 thr) per token
// =====================================================================
__global__ __launch_bounds__(256)
void layernorm_kernel(const float* __restrict__ x, const float* __restrict__ g,
                      const float* __restrict__ bta, float* __restrict__ out)
{
    __shared__ float red[256];
    const int t = blockIdx.x, tid = threadIdx.x;
    const float* xr = x + (size_t)t * DIM;
    float vals[4], s = 0.f, ss = 0.f;
    #pragma unroll
    for (int i = 0; i < 4; ++i) {
        float v = xr[tid + i * 256];
        vals[i] = v; s += v; ss += v * v;
    }
    red[tid] = s; __syncthreads();
    for (int o = 128; o > 0; o >>= 1) { if (tid < o) red[tid] += red[tid + o]; __syncthreads(); }
    const float mu = red[0] * (1.0f / DIM);
    __syncthreads();
    red[tid] = ss; __syncthreads();
    for (int o = 128; o > 0; o >>= 1) { if (tid < o) red[tid] += red[tid + o]; __syncthreads(); }
    const float var  = red[0] * (1.0f / DIM) - mu * mu;
    const float rstd = rsqrtf(var + 1e-5f);
    float* orow = out + (size_t)t * DIM;
    #pragma unroll
    for (int i = 0; i < 4; ++i) {
        int d = tid + i * 256;
        orow[d] = (vals[i] - mu) * rstd * g[d] + bta[d];
    }
}

// =====================================================================
// Flash attention: 1 wave per (b, h, 16-query tile). qkv: [T, 3*D] fp32.
// =====================================================================
__global__ __launch_bounds__(256)
void attention_kernel(const float* __restrict__ qkv, float* __restrict__ ctx)
{
    __shared__ __align__(16) __bf16 Pbuf[8][16][40];  // per-wave P tile (16x32)

    const int tid  = threadIdx.x;
    const int wave = tid >> 5, lane = tid & 31;
    const int lo   = lane & 15, hi = lane >> 4;
    const int w  = blockIdx.x * 8 + wave;       // 512 blocks * 8 waves = 4096
    const int bh = w >> 6, qt = w & 63;         // 64 q-tiles of 16 rows
    const int b  = bh >> 4, hh = bh & 15;
    const int rstride = 3 * DIM;
    const float scale = 0.125f;                 // 1/sqrt(64)

    // ---- load Q as two A-fragments (DK=64 -> two K=32 steps) ----
    FragA qf[2];
    {
        const float* qp = qkv + (size_t)(b * SEQ + qt * 16 + lo) * rstride + hh * DKH;
        #pragma unroll
        for (int f = 0; f < 2; ++f) {
            load8_cvt(&qf[f].h[0], qp + f * 32 + hi * 8);        // K {0..7} or {8..15}
            load8_cvt(&qf[f].h[8], qp + f * 32 + 16 + hi * 8);   // K {16..23} or {24..31}
        }
    }

    float mx[8], lsum[8];
    v8f acc[4] = {};
    #pragma unroll
    for (int r = 0; r < 8; ++r) { mx[r] = -1e30f; lsum[r] = 0.f; }

    for (int j = 0; j < SEQ / 32; ++j) {
        // ---- scores for 32 keys as two 16x16 D tiles ----
        v8f s[2];
        #pragma unroll
        for (int jj = 0; jj < 2; ++jj) {
            const float* kp = qkv + (size_t)(b * SEQ + j * 32 + jj * 16 + lo) * rstride
                            + DIM + hh * DKH;
            FragA kf0, kf1;                       // B frags: lane n = key, K split like Q
            load8_cvt(&kf0.h[0], kp + hi * 8);
            load8_cvt(&kf0.h[8], kp + 16 + hi * 8);
            load8_cvt(&kf1.h[0], kp + 32 + hi * 8);
            load8_cvt(&kf1.h[8], kp + 32 + 16 + hi * 8);
            v8f z = {};
            z = __builtin_amdgcn_wmma_f32_16x16x32_bf16(false, qf[0].v, false, kf0.v, (short)0, z, false, false);
            z = __builtin_amdgcn_wmma_f32_16x16x32_bf16(false, qf[1].v, false, kf1.v, (short)0, z, false, false);
            s[jj] = z;
        }
        // ---- online softmax (rows live in 16-lane halves) ----
        float p0[8], p1[8];
        #pragma unroll
        for (int r = 0; r < 8; ++r) {
            float v0 = s[0][r] * scale, v1 = s[1][r] * scale;
            float m = fmaxf(v0, v1);
            for (int msk = 1; msk < 16; msk <<= 1) m = fmaxf(m, __shfl_xor(m, msk, 32));
            const float nm   = fmaxf(mx[r], m);
            const float corr = expf(mx[r] - nm);
            const float e0 = expf(v0 - nm), e1 = expf(v1 - nm);
            float ps = e0 + e1;
            for (int msk = 1; msk < 16; msk <<= 1) ps += __shfl_xor(ps, msk, 32);
            lsum[r] = lsum[r] * corr + ps;
            mx[r] = nm; p0[r] = e0; p1[r] = e1;
            #pragma unroll
            for (int tv = 0; tv < 4; ++tv) acc[tv][r] *= corr;
        }
        // ---- P: C/D layout -> LDS -> A-fragment layout ----
        __syncthreads();
        #pragma unroll
        for (int r = 0; r < 8; ++r) {
            Pbuf[wave][hi * 8 + r][lo]      = (__bf16)p0[r];
            Pbuf[wave][hi * 8 + r][16 + lo] = (__bf16)p1[r];
        }
        __syncthreads();
        FragA pf;
        {
            const __bf16* rp = &Pbuf[wave][lo][0];
            pf.q[0] = *(const uint4*)(rp + hi * 8);
            pf.q[1] = *(const uint4*)(rp + 16 + hi * 8);
        }
        // ---- ctx += P x V  (4 n-tiles over DK=64) ----
        const float* vbase = qkv + (size_t)(b * SEQ + j * 32 + hi * 8) * rstride
                           + 2 * DIM + hh * DKH;
        #pragma unroll
        for (int tv = 0; tv < 4; ++tv) {
            FragA vf;                               // B frag: lane n = dk column, K = kv rows
            const float* vp = vbase + tv * 16 + lo;
            #pragma unroll
            for (int i = 0; i < 16; ++i) vf.h[i] = (__bf16)vp[(i + (i >= 8 ? 8 : 0)) * rstride];
            acc[tv] = __builtin_amdgcn_wmma_f32_16x16x32_bf16(
                false, pf.v, false, vf.v, (short)0, acc[tv], false, false);
        }
    }
    // ---- normalize and write ctx [T, D] ----
    #pragma unroll
    for (int tv = 0; tv < 4; ++tv)
        #pragma unroll
        for (int r = 0; r < 8; ++r) {
            const int m = hi * 8 + r;
            ctx[(size_t)(b * SEQ + qt * 16 + m) * DIM + hh * DKH + tv * 16 + lo] =
                acc[tv][r] / lsum[r];
        }
}

// =====================================================================
// Router: 1 wave per token. logits, top-2 normalized gates, entropy.
// =====================================================================
__global__ __launch_bounds__(256)
void router_kernel(const float* __restrict__ nx, const float* __restrict__ Wg,
                   float* __restrict__ logits_out, float* __restrict__ ew,
                   float* __restrict__ ent)
{
    const int tid = threadIdx.x;
    const int wave = tid >> 5, lane = tid & 31;
    const int t = blockIdx.x * 8 + wave;
    float a[8] = {};
    const float* xr = nx + (size_t)t * DIM;
    for (int d = lane; d < DIM; d += 32) {
        const float xv = xr[d];
        #pragma unroll
        for (int e = 0; e < 8; ++e) a[e] += xv * Wg[e * DIM + d];
    }
    #pragma unroll
    for (int e = 0; e < 8; ++e)
        for (int msk = 1; msk < 32; msk <<= 1) a[e] += __shfl_xor(a[e], msk, 32);
    if (lane == 0) {
        float mxv = a[0];
        #pragma unroll
        for (int e = 1; e < 8; ++e) mxv = fmaxf(mxv, a[e]);
        float p[8], s = 0.f;
        #pragma unroll
        for (int e = 0; e < 8; ++e) { p[e] = expf(a[e] - mxv); s += p[e]; }
        const float inv = 1.f / s;
        float H = 0.f;
        #pragma unroll
        for (int e = 0; e < 8; ++e) {
            p[e] *= inv;
            logits_out[(size_t)t * 8 + e] = a[e];
            H -= p[e] * logf(fmaxf(p[e], 1e-6f));
        }
        ent[t] = H;
        int i1 = 0;
        #pragma unroll
        for (int e = 1; e < 8; ++e) if (p[e] > p[i1]) i1 = e;
        int i2 = (i1 == 0) ? 1 : 0;
        #pragma unroll
        for (int e = 0; e < 8; ++e) if (e != i1 && p[e] > p[i2]) i2 = e;
        const float wsum = p[i1] + p[i2];
        #pragma unroll
        for (int e = 0; e < 8; ++e) ew[(size_t)t * 8 + e] = 0.f;
        ew[(size_t)t * 8 + i1] = p[i1] / wsum;
        ew[(size_t)t * 8 + i2] = p[i2] / wsum;
    }
}

__global__ __launch_bounds__(256)
void entropy_reduce_kernel(const float* __restrict__ ent, float* __restrict__ out)
{
    __shared__ float red[256];
    float s = 0.f;
    for (int i = threadIdx.x; i < TOK; i += 256) s += ent[i];
    red[threadIdx.x] = s; __syncthreads();
    for (int o = 128; o > 0; o >>= 1) {
        if (threadIdx.x < o) red[threadIdx.x] += red[threadIdx.x + o];
        __syncthreads();
    }
    if (threadIdx.x == 0) out[0] = red[0] / (float)TOK;
}

// =====================================================================
extern "C" void kernel_launch(void* const* d_in, const int* in_sizes, int n_in,
                              void* d_out, int out_size, void* d_ws, size_t ws_size,
                              hipStream_t stream)
{
    const float* x    = (const float*)d_in[0];
    const float* Wqkv = (const float*)d_in[1];
    const float* Wo   = (const float*)d_in[2];
    const float* ln1g = (const float*)d_in[3];
    const float* ln1b = (const float*)d_in[4];
    const float* ln2g = (const float*)d_in[5];
    const float* ln2b = (const float*)d_in[6];
    const float* Wg   = (const float*)d_in[7];
    const float* W1   = (const float*)d_in[8];
    const float* W2   = (const float*)d_in[9];
    const float* rs   = (const float*)d_in[10];
    float* out = (float*)d_out;

    const size_t TD = (size_t)TOK * DIM;           // 4 Mi
    float* ws   = (float*)d_ws;
    float* nx1  = ws;                              // [0, TD)       (dead after qkv gemm)
    float* qkvb = ws + TD;                         // [TD, 4TD)     (dead after attention)
    float* ctx  = nx1;                             // reuse [0, TD)
    float* x1   = ws + TD;                         // reuse [TD, 2TD)
    float* nx2  = ws + 2 * TD;                     // [2TD, 3TD)
    float* h    = ws + 3 * TD;                     // [3TD, 5TD)    (T x DFF)
    float* ew   = ws + 5 * TD;                     // [5TD, 5TD + T*8)
    float* ent  = ws + 5 * TD + (size_t)TOK * 8;   // T floats

    // 1) LN1
    layernorm_kernel<<<TOK, 256, 0, stream>>>(x, ln1g, ln1b, nx1);
    // 2) fused QKV projection: [T,3D] = nx1 @ Wqkv^T
    gemm_bf16_kernel<<<dim3(3 * DIM / 128, TOK / 64), 256, 0, stream>>>(
        nx1, Wqkv, qkvb, nullptr, nullptr, nullptr, nullptr, TOK, 3 * DIM, DIM, 0);
    // 3) flash attention -> ctx [T,D]
    attention_kernel<<<(BATCH * NH * (SEQ / 16)) / 8, 256, 0, stream>>>(qkvb, ctx);
    // 4) output proj + residual: x1 = x + (ctx @ Wo^T)*rs ; also seed d_out x-region
    gemm_bf16_kernel<<<dim3(DIM / 128, TOK / 64), 256, 0, stream>>>(
        ctx, Wo, x1, out, x, nullptr, rs, TOK, DIM, DIM, 2);
    // 5) LN2
    layernorm_kernel<<<TOK, 256, 0, stream>>>(x1, ln2g, ln2b, nx2);
    // 6) router: logits -> d_out, gates -> ew, per-token entropy -> ent
    router_kernel<<<TOK / 8, 256, 0, stream>>>(nx2, Wg, out + TD, ew, ent);
    entropy_reduce_kernel<<<1, 256, 0, stream>>>(ent, out + TD + (size_t)TOK * NE);
    // 7) experts (serialized on stream -> deterministic accumulation into d_out)
    for (int e = 0; e < NE; ++e) {
        gemm_bf16_kernel<<<dim3(DFF / 128, TOK / 64), 256, 0, stream>>>(
            nx2, W1 + (size_t)e * DFF * DIM, h,
            nullptr, nullptr, nullptr, nullptr, TOK, DFF, DIM, 1);
        gemm_bf16_kernel<<<dim3(DIM / 128, TOK / 64), 256, 0, stream>>>(
            h, W2 + (size_t)e * DIM * DFF, out,
            nullptr, nullptr, ew + e, rs, TOK, DIM, DFF, 3);
    }
    (void)in_sizes; (void)n_in; (void)out_size; (void)ws_size;
}
